// ParallelExperts_5935644803776
// MI455X (gfx1250) — matrix operand order repaired and checked by reading, not compiled, not run on hardware
//
#include <hip/hip_runtime.h>

typedef __attribute__((ext_vector_type(16))) __bf16 v16bf;
typedef __attribute__((ext_vector_type(8)))  __bf16 v8bf;
typedef __attribute__((ext_vector_type(4)))  __bf16 v4bf;
typedef __attribute__((ext_vector_type(8)))  float  v8f;

#define TOKENS 8192
#define D_IN   2048
#define D_OUT  8192
#define NEXP   8

#define BM 128
#define BN 128
#define BK 32
#define LDA 40   // bf16 elems per LDS row (BK + 8 pad), multiple of 8 for 16B-aligned frag loads
#define LDB 40

#define SHUF16(a, b) __builtin_shufflevector(a, b, 0,1,2,3,4,5,6,7,8,9,10,11,12,13,14,15)

__global__ __launch_bounds__(256)
void moe_grouped_gemm_bf16x3(const float* __restrict__ A,
                             const float* __restrict__ W,
                             const long long* __restrict__ esz,
                             float* __restrict__ out)
{
    __shared__ __bf16 sAh[BM * LDA];
    __shared__ __bf16 sAl[BM * LDA];
    __shared__ __bf16 sBh[BN * LDB];
    __shared__ __bf16 sBl[BN * LDB];

    const int tid  = threadIdx.x;
    const int lane = tid & 31;
    const int wid  = tid >> 5;

    const int rowBase = blockIdx.x * BM;   // token rows (== output rows)
    const int colBase = blockIdx.y * BN;   // output cols

    // ---- expert lookup: row tiles never span experts (all split boundaries
    // are multiples of 128 for the static EXPERT_SIZES) ----
    int e = 0;
    long long cum = 0;
    for (int i = 0; i < NEXP - 1; ++i) {
        cum += esz[i];
        if ((long long)rowBase >= cum) e = i + 1;
    }
    const float* __restrict__ Wb = W + (size_t)e * D_IN * D_OUT;

    // 8 waves: 4 along M x 2 along N; each wave owns 32(M) x 64(N)
    const int waveM = (wid & 3) * 32;
    const int waveN = (wid >> 2) * 64;

    v8f acc[2][4];
    for (int mt = 0; mt < 2; ++mt)
        for (int nt = 0; nt < 4; ++nt)
            acc[mt][nt] = (v8f)(0.0f);

    float4 ra[4], rb[4];   // staging registers (software pipeline)

    // ---------------- fetch chunk k0 into registers ----------------
    auto fetch = [&](int k0) {
#pragma unroll
        for (int i = 0; i < 4; ++i) {
            int idx = tid + i * 256;
            int r  = idx >> 3;            // 0..127   (A row in tile)
            int kq = (idx & 7) * 4;       // 0..28    (k quad)
            ra[i] = *(const float4*)(A + (size_t)(rowBase + r) * D_IN + k0 + kq);
            int kk = idx >> 5;            // 0..31    (B k row)
            int nq = (idx & 31) * 4;      // 0..124   (n quad, coalesced)
            rb[i] = *(const float4*)(Wb + (size_t)(k0 + kk) * D_OUT + colBase + nq);
        }
    };

    // ---------------- registers -> LDS (bf16 hi/lo split) ----------------
    auto stage = [&]() {
#pragma unroll
        for (int i = 0; i < 4; ++i) {
            int idx = tid + i * 256;
            // A: row-major [m][k]
            int r  = idx >> 3;
            int kq = (idx & 7) * 4;
            float4 v = ra[i];
            __bf16 h0 = (__bf16)v.x, h1 = (__bf16)v.y, h2 = (__bf16)v.z, h3 = (__bf16)v.w;
            v4bf hv; hv[0] = h0; hv[1] = h1; hv[2] = h2; hv[3] = h3;
            v4bf lv;
            lv[0] = (__bf16)(v.x - (float)h0);
            lv[1] = (__bf16)(v.y - (float)h1);
            lv[2] = (__bf16)(v.z - (float)h2);
            lv[3] = (__bf16)(v.w - (float)h3);
            *(v4bf*)(&sAh[r * LDA + kq]) = hv;   // ds_store_b64
            *(v4bf*)(&sAl[r * LDA + kq]) = lv;
            // B: transpose into col-major-by-N [n][k]
            int kk = idx >> 5;
            int nq = (idx & 31) * 4;
            float4 w = rb[i];
            float wf[4] = { w.x, w.y, w.z, w.w };
#pragma unroll
            for (int j = 0; j < 4; ++j) {
                __bf16 h = (__bf16)wf[j];
                sBh[(nq + j) * LDB + kk] = h;
                sBl[(nq + j) * LDB + kk] = (__bf16)(wf[j] - (float)h);
            }
        }
    };

    fetch(0);
    stage();

    const int mrow = lane & 15;
    const int kha  = (lane >> 4) * 8;    // A frag: K = kha..kha+7 and kha+16..kha+23
    const int khb  = (lane >> 4) * 8;    // B frag: K = khb..khb+7 and khb+16..khb+23

    for (int k0 = 0; k0 < D_IN; k0 += BK) {
        __syncthreads();                       // LDS tiles for k0 ready
        const bool more = (k0 + BK) < D_IN;
        if (more) fetch(k0 + BK);              // global prefetch overlaps WMMA

        // ---- load fragments (ds_load_b128 pairs, 16B aligned) ----
        v16bf afh[2], afl[2];
#pragma unroll
        for (int mt = 0; mt < 2; ++mt) {
            const __bf16* ph = &sAh[(waveM + mt * 16 + mrow) * LDA + kha];
            const __bf16* pl = &sAl[(waveM + mt * 16 + mrow) * LDA + kha];
            v8bf a0 = *(const v8bf*)(ph);
            v8bf a1 = *(const v8bf*)(ph + 16);
            afh[mt] = SHUF16(a0, a1);
            v8bf b0 = *(const v8bf*)(pl);
            v8bf b1 = *(const v8bf*)(pl + 16);
            afl[mt] = SHUF16(b0, b1);
        }
        v16bf bfh[4], bfl[4];
#pragma unroll
        for (int nt = 0; nt < 4; ++nt) {
            const __bf16* ph = &sBh[(waveN + nt * 16 + mrow) * LDB + khb];
            const __bf16* pl = &sBl[(waveN + nt * 16 + mrow) * LDB + khb];
            v8bf a0 = *(const v8bf*)(ph);
            v8bf a1 = *(const v8bf*)(ph + 16);
            bfh[nt] = SHUF16(a0, a1);
            v8bf b0 = *(const v8bf*)(pl);
            v8bf b1 = *(const v8bf*)(pl + 16);
            bfl[nt] = SHUF16(b0, b1);
        }

        // ---- bf16x3: hi*hi + hi*lo + lo*hi  (lo*lo term ~2^-16 relative, dropped) ----
#pragma unroll
        for (int mt = 0; mt < 2; ++mt)
#pragma unroll
            for (int nt = 0; nt < 4; ++nt) {
                acc[mt][nt] = __builtin_amdgcn_wmma_f32_16x16x32_bf16(
                    false, afh[mt], false, bfh[nt], (short)0, acc[mt][nt], false, false);
                acc[mt][nt] = __builtin_amdgcn_wmma_f32_16x16x32_bf16(
                    false, afh[mt], false, bfl[nt], (short)0, acc[mt][nt], false, false);
                acc[mt][nt] = __builtin_amdgcn_wmma_f32_16x16x32_bf16(
                    false, afl[mt], false, bfh[nt], (short)0, acc[mt][nt], false, false);
            }

        __syncthreads();                       // everyone done reading LDS
        if (more) stage();                     // write next chunk to LDS
    }

    // ---- epilogue: C layout -> global (half-wave writes 16 contiguous floats) ----
    const int nl = lane & 15;
    const int mh = (lane >> 4) * 8;
#pragma unroll
    for (int mt = 0; mt < 2; ++mt)
#pragma unroll
        for (int nt = 0; nt < 4; ++nt) {
            int row0 = rowBase + waveM + mt * 16 + mh;
            int col  = colBase + waveN + nt * 16 + nl;
#pragma unroll
            for (int r = 0; r < 8; ++r)
                out[(size_t)(row0 + r) * D_OUT + col] = acc[mt][nt][r];
        }
}

extern "C" void kernel_launch(void* const* d_in, const int* in_sizes, int n_in,
                              void* d_out, int out_size, void* d_ws, size_t ws_size,
                              hipStream_t stream) {
    (void)in_sizes; (void)n_in; (void)out_size; (void)d_ws; (void)ws_size;
    const float*     A   = (const float*)d_in[0];     // [8192, 2048] f32
    const float*     W   = (const float*)d_in[1];     // [8, 2048, 8192] f32
    const long long* esz = (const long long*)d_in[2]; // [8] int64
    float*           out = (float*)d_out;             // [8192, 8192] f32

    dim3 grid(TOKENS / BM, D_OUT / BN);  // 64 x 64 tiles
    dim3 block(256);                     // 8 wave32s
    hipLaunchKernelGGL(moe_grouped_gemm_bf16x3, grid, block, 0, stream, A, W, esz, out);
}
